// Attention_39436389712086
// MI455X (gfx1250) — matrix operand order repaired and checked by reading, not compiled, not run on hardware
//
#include <hip/hip_runtime.h>

// ---------------------------------------------------------------------------
// Attention block for MI455X (gfx1250, wave32, WMMA 16x16x32 f16->f32)
// B=8, N=1024, DIM=768, H=12, HEAD_DIM=64
// ---------------------------------------------------------------------------

typedef _Float16 half_t;
typedef __attribute__((ext_vector_type(16))) _Float16 v16h;
typedef __attribute__((ext_vector_type(8)))  _Float16 v8h;
typedef __attribute__((ext_vector_type(8)))  float    v8f;

#define BATCH    8
#define SEQ      1024
#define DIM      768
#define NHEAD    12
#define HDIM     64
#define QKVDIM   (3 * DIM)          // 2304
#define ROWS     (BATCH * SEQ)      // 8192
#define SCALE    0.125f             // HDIM^-0.5

// ---------------------------------------------------------------------------
// WMMA helper: D = A(16x32 f16) * B(32x16 f16) + C(16x16 f32)
// ---------------------------------------------------------------------------
__device__ __forceinline__ v8f wmma_f16(v16h a, v16h b, v8f c) {
    return __builtin_amdgcn_wmma_f32_16x16x32_f16(
        /*neg_a=*/false, a, /*neg_b=*/false, b,
        /*c_mod=*/(short)0, c, /*reuse_a=*/false, /*reuse_b=*/false);
}

// A fragment (16x32, f16) from row-major memory: base = &tile(0, k0), ld = row stride.
// ISA layout: lane r (r = lane&15) holds row r; chunks at k = half*8 and 16+half*8.
__device__ __forceinline__ v16h load_a_frag(const half_t* __restrict__ base, int ld) {
    const int lane = threadIdx.x & 31;
    const int r  = lane & 15;
    const int hf = lane >> 4;
    const half_t* p = base + (size_t)r * ld + hf * 8;
    union { v16h v; v8h h[2]; } u;
    u.h[0] = *(const v8h*)(p);
    u.h[1] = *(const v8h*)(p + 16);
    return u.v;
}

// B fragment (32x16, f16) where element (k, n) lives at base[n*ldn + k]
// (i.e. each output column n is k-contiguous in memory).
// ISA layout: lane holds column n = lane&15; chunks at k = half*8 and 16+half*8.
__device__ __forceinline__ v16h load_b_frag(const half_t* __restrict__ base, int ldn) {
    const int lane = threadIdx.x & 31;
    const int n  = lane & 15;
    const int hf = lane >> 4;
    const half_t* p = base + (size_t)n * ldn + hf * 8;
    union { v16h v; v8h h[2]; } u;
    u.h[0] = *(const v8h*)(p);
    u.h[1] = *(const v8h*)(p + 16);
    return u.v;
}

// A fragment built from f32 data in LDS (softmax probabilities).
__device__ __forceinline__ v16h load_a_frag_lds_f32(const float* __restrict__ s,
                                                    int ld, int j0) {
    const int lane = threadIdx.x & 31;
    const int r  = lane & 15;
    const int hf = lane >> 4;
    const float* p = s + r * ld + j0 + hf * 8;
    v16h out;
#pragma unroll
    for (int e = 0; e < 8; ++e) out[e]     = (half_t)p[e];
#pragma unroll
    for (int e = 0; e < 8; ++e) out[8 + e] = (half_t)p[16 + e];
    return out;
}

// ---------------------------------------------------------------------------
// f32 -> f16 conversion
// ---------------------------------------------------------------------------
__global__ void cvt_f32_to_f16(const float* __restrict__ src,
                               half_t* __restrict__ dst, int n) {
    int i = blockIdx.x * blockDim.x + threadIdx.x;
    if (i < n) dst[i] = (half_t)src[i];
}

// ---------------------------------------------------------------------------
// QKV GEMM: qkv[row, e] = sum_d x[row, d] * w_qkv[e, d]
// One wave computes a 32(M) x 64(E) tile: 2 A frags, 4 B frags, 8 WMMAs / k-step.
// block = 128 (4 waves along E), grid = (ROWS/32, QKVDIM/256).
// Epilogue scatters into Q/K as [b, h, n, d] (f16) and V as [b, h, d, n] (f16).
// ---------------------------------------------------------------------------
__global__ void qkv_gemm_kernel(const half_t* __restrict__ xh,
                                const half_t* __restrict__ wh,
                                half_t* __restrict__ q,
                                half_t* __restrict__ k,
                                half_t* __restrict__ v) {
    const int wave = threadIdx.x >> 5;
    const int e0   = blockIdx.y * 256 + wave * 64;   // 64-aligned => single (qkv,h)
    const int row0 = blockIdx.x * 32;

    const half_t* abase0 = xh + (size_t)row0 * DIM;
    const half_t* abase1 = xh + (size_t)(row0 + 16) * DIM;

    v8f accA[4] = {};   // rows row0 .. row0+15
    v8f accB[4] = {};   // rows row0+16 .. row0+31

#pragma unroll 2
    for (int kk = 0; kk < DIM; kk += 32) {
        // Issue all fragment loads for this k-step before any WMMA.
        v16h a0 = load_a_frag(abase0 + kk, DIM);
        v16h a1 = load_a_frag(abase1 + kk, DIM);
        v16h b0 = load_b_frag(wh + (size_t)(e0 +  0) * DIM + kk, DIM);
        v16h b1 = load_b_frag(wh + (size_t)(e0 + 16) * DIM + kk, DIM);
        v16h b2 = load_b_frag(wh + (size_t)(e0 + 32) * DIM + kk, DIM);
        v16h b3 = load_b_frag(wh + (size_t)(e0 + 48) * DIM + kk, DIM);

        // Pull next k-step toward the WGP while the WMMAs execute.
        __builtin_prefetch(abase0 + kk + 32, 0, 1);
        __builtin_prefetch(wh + (size_t)e0 * DIM + kk + 32, 0, 1);

        accA[0] = wmma_f16(a0, b0, accA[0]);
        accA[1] = wmma_f16(a0, b1, accA[1]);
        accA[2] = wmma_f16(a0, b2, accA[2]);
        accA[3] = wmma_f16(a0, b3, accA[3]);
        accB[0] = wmma_f16(a1, b0, accB[0]);
        accB[1] = wmma_f16(a1, b1, accB[1]);
        accB[2] = wmma_f16(a1, b2, accB[2]);
        accB[3] = wmma_f16(a1, b3, accB[3]);
    }

    const int lane = threadIdx.x & 31;
    const int nn   = lane & 15;
    const int hf   = lane >> 4;
    const int qkv_idx = e0 / DIM;              // 0=Q 1=K 2=V (wave-uniform)
    const int h       = (e0 % DIM) / HDIM;

#pragma unroll
    for (int half16 = 0; half16 < 2; ++half16) {
        const v8f* acc = half16 ? accB : accA;
#pragma unroll
        for (int r = 0; r < 8; ++r) {
            const int m    = r + 8 * hf + 16 * half16;
            const int grow = row0 + m;
            const int b    = grow >> 10;           // / SEQ
            const int npos = grow & (SEQ - 1);
            const size_t bh = (size_t)b * NHEAD + h;
#pragma unroll
            for (int t = 0; t < 4; ++t) {
                const int d = t * 16 + nn;         // 0..63 within head
                const half_t val = (half_t)acc[t][r];
                if (qkv_idx == 0) {
                    q[(bh * SEQ + npos) * HDIM + d] = val;
                } else if (qkv_idx == 1) {
                    k[(bh * SEQ + npos) * HDIM + d] = val;
                } else {
                    v[(bh * HDIM + d) * SEQ + npos] = val;   // transposed
                }
            }
        }
    }
}

// ---------------------------------------------------------------------------
// Attention core: one wave per (b, h, 16-row block).
// Pass 1: S = scale * Q K^T into LDS (16 x 1024 f32); 32 columns / iteration.
// Pass 2: row softmax in LDS (unnormalized exp; keep 1/rowsum in registers).
// Pass 3: O = P V (V transposed), scale rows by 1/rowsum, store ctx f16
//         in [b, n, h*64+d] layout for the projection GEMM.
// ---------------------------------------------------------------------------
__global__ void attention_kernel(const half_t* __restrict__ q,
                                 const half_t* __restrict__ k,
                                 const half_t* __restrict__ v,
                                 half_t* __restrict__ ctx) {
    extern __shared__ float smem[];            // 16 * 1024 floats = 64 KB
    const int LDP = SEQ;

    const int bid = blockIdx.x;
    const int it  = bid & 63;                  // row-block within sequence
    const int bh  = bid >> 6;                  // b*NHEAD + h
    const int i0  = it * 16;

    const half_t* qb = q + (size_t)bh * SEQ * HDIM;
    const half_t* kb = k + (size_t)bh * SEQ * HDIM;
    const half_t* vb = v + (size_t)bh * HDIM * SEQ;   // [d][n]

    const int lane = threadIdx.x & 31;
    const int nn   = lane & 15;
    const int hf   = lane >> 4;

    // Q fragments stay resident for the whole wave.
    const v16h aq0 = load_a_frag(qb + (size_t)i0 * HDIM + 0,  HDIM);
    const v16h aq1 = load_a_frag(qb + (size_t)i0 * HDIM + 32, HDIM);

    // ---- Pass 1: scores (two 16-column tiles per iteration) ----
    for (int j0 = 0; j0 < SEQ; j0 += 32) {
        v16h bk0 = load_b_frag(kb + (size_t)(j0 +  0) * HDIM + 0,  HDIM);
        v16h bk1 = load_b_frag(kb + (size_t)(j0 +  0) * HDIM + 32, HDIM);
        v16h bk2 = load_b_frag(kb + (size_t)(j0 + 16) * HDIM + 0,  HDIM);
        v16h bk3 = load_b_frag(kb + (size_t)(j0 + 16) * HDIM + 32, HDIM);
        v8f c0 = {}, c1 = {};
        c0 = wmma_f16(aq0, bk0, c0);
        c0 = wmma_f16(aq1, bk1, c0);
        c1 = wmma_f16(aq0, bk2, c1);
        c1 = wmma_f16(aq1, bk3, c1);
#pragma unroll
        for (int r = 0; r < 8; ++r) {
            const int m = r + 8 * hf;
            smem[m * LDP + j0 + nn]      = c0[r] * SCALE;
            smem[m * LDP + j0 + 16 + nn] = c1[r] * SCALE;
        }
    }
    __syncthreads();

    // ---- Pass 2: softmax (2 lanes per row; lane handles 512 columns) ----
    float inv_sum;
    {
        const int row = lane & 15;
        float* srow = smem + row * LDP;
        const int jb = hf * 512;
        float mx = -3.0e38f;
        for (int j = 0; j < 512; ++j) mx = fmaxf(mx, srow[jb + j]);
        mx = fmaxf(mx, __shfl_xor(mx, 16, 32));
        float sum = 0.0f;
        for (int j = 0; j < 512; ++j) {
            const float e = __expf(srow[jb + j] - mx);
            srow[jb + j] = e;
            sum += e;
        }
        sum += __shfl_xor(sum, 16, 32);
        inv_sum = 1.0f / sum;                  // every lane: 1/rowsum of row=lane&15
    }
    __syncthreads();

    // ---- Pass 3: O = P V ----
    v8f o[4] = {};
    for (int j0 = 0; j0 < SEQ; j0 += 32) {
        v16h bv0 = load_b_frag(vb + (size_t)( 0) * SEQ + j0, SEQ);
        v16h bv1 = load_b_frag(vb + (size_t)(16) * SEQ + j0, SEQ);
        v16h bv2 = load_b_frag(vb + (size_t)(32) * SEQ + j0, SEQ);
        v16h bv3 = load_b_frag(vb + (size_t)(48) * SEQ + j0, SEQ);
        v16h ap  = load_a_frag_lds_f32(smem, LDP, j0);
        o[0] = wmma_f16(ap, bv0, o[0]);
        o[1] = wmma_f16(ap, bv1, o[1]);
        o[2] = wmma_f16(ap, bv2, o[2]);
        o[3] = wmma_f16(ap, bv3, o[3]);
    }

    // ---- Epilogue: normalize rows, store ctx[b, n, h*64 + d] ----
    const int b = bh / NHEAD;
    const int h = bh % NHEAD;
#pragma unroll
    for (int r = 0; r < 8; ++r) {
        const int m   = r + 8 * hf;
        const float inv = __shfl(inv_sum, m, 32);    // lane m holds row m's 1/sum
        const int npos  = i0 + m;
#pragma unroll
        for (int t = 0; t < 4; ++t) {
            ctx[((size_t)b * SEQ + npos) * DIM + h * HDIM + t * 16 + nn] =
                (half_t)(o[t][r] * inv);
        }
    }
}

// ---------------------------------------------------------------------------
// Output projection: out[row, e] = sum_d ctx[row, d] * w_proj[e, d] + b_proj[e]
// One wave per 32(M) x 64(E) tile; block = 128, grid = (ROWS/32, DIM/256).
// ---------------------------------------------------------------------------
__global__ void proj_gemm_kernel(const half_t* __restrict__ ctx,
                                 const half_t* __restrict__ wh,
                                 const float* __restrict__ bias,
                                 float* __restrict__ out) {
    const int wave = threadIdx.x >> 5;
    const int e0   = blockIdx.y * 256 + wave * 64;
    const int row0 = blockIdx.x * 32;

    const half_t* abase0 = ctx + (size_t)row0 * DIM;
    const half_t* abase1 = ctx + (size_t)(row0 + 16) * DIM;

    v8f accA[4] = {};
    v8f accB[4] = {};

#pragma unroll 2
    for (int kk = 0; kk < DIM; kk += 32) {
        v16h a0 = load_a_frag(abase0 + kk, DIM);
        v16h a1 = load_a_frag(abase1 + kk, DIM);
        v16h b0 = load_b_frag(wh + (size_t)(e0 +  0) * DIM + kk, DIM);
        v16h b1 = load_b_frag(wh + (size_t)(e0 + 16) * DIM + kk, DIM);
        v16h b2 = load_b_frag(wh + (size_t)(e0 + 32) * DIM + kk, DIM);
        v16h b3 = load_b_frag(wh + (size_t)(e0 + 48) * DIM + kk, DIM);

        __builtin_prefetch(abase0 + kk + 32, 0, 1);
        __builtin_prefetch(wh + (size_t)e0 * DIM + kk + 32, 0, 1);

        accA[0] = wmma_f16(a0, b0, accA[0]);
        accA[1] = wmma_f16(a0, b1, accA[1]);
        accA[2] = wmma_f16(a0, b2, accA[2]);
        accA[3] = wmma_f16(a0, b3, accA[3]);
        accB[0] = wmma_f16(a1, b0, accB[0]);
        accB[1] = wmma_f16(a1, b1, accB[1]);
        accB[2] = wmma_f16(a1, b2, accB[2]);
        accB[3] = wmma_f16(a1, b3, accB[3]);
    }

    const int lane = threadIdx.x & 31;
    const int nn   = lane & 15;
    const int hf   = lane >> 4;
#pragma unroll
    for (int half16 = 0; half16 < 2; ++half16) {
        const v8f* acc = half16 ? accB : accA;
#pragma unroll
        for (int r = 0; r < 8; ++r) {
            const int m = r + 8 * hf + 16 * half16;
#pragma unroll
            for (int t = 0; t < 4; ++t) {
                const int e = e0 + t * 16 + nn;
                out[(size_t)(row0 + m) * DIM + e] = acc[t][r] + bias[e];
            }
        }
    }
}

// ---------------------------------------------------------------------------
// Host-side launcher
// ---------------------------------------------------------------------------
extern "C" void kernel_launch(void* const* d_in, const int* in_sizes, int n_in,
                              void* d_out, int out_size, void* d_ws, size_t ws_size,
                              hipStream_t stream) {
    const float* x      = (const float*)d_in[0];   // [8, 1024, 768]
    const float* w_qkv  = (const float*)d_in[1];   // [2304, 768]
    const float* w_proj = (const float*)d_in[2];   // [768, 768]
    const float* b_proj = (const float*)d_in[3];   // [768]
    float* out = (float*)d_out;                    // [8, 1024, 768]

    // Workspace carve-up (f16 buffers), each region 256B-aligned by construction.
    char* ws = (char*)d_ws;
    const size_t xh_bytes   = (size_t)ROWS * DIM * sizeof(half_t);      // 12.58 MB
    const size_t wqkv_bytes = (size_t)QKVDIM * DIM * sizeof(half_t);    //  3.54 MB
    const size_t wprj_bytes = (size_t)DIM * DIM * sizeof(half_t);       //  1.18 MB
    const size_t head_bytes = (size_t)BATCH * NHEAD * SEQ * HDIM * sizeof(half_t);

    half_t* xh    = (half_t*)ws;            ws += xh_bytes;
    half_t* wqkvh = (half_t*)ws;            ws += wqkv_bytes;
    half_t* wprjh = (half_t*)ws;            ws += wprj_bytes;
    half_t* qbuf  = (half_t*)ws;            ws += head_bytes;
    half_t* kbuf  = (half_t*)ws;            ws += head_bytes;
    half_t* vbuf  = (half_t*)ws;            ws += head_bytes;   // transposed [b,h,d,n]
    half_t* ctx   = (half_t*)ws;            ws += xh_bytes;

    // 1) Precision conversion (one-shot, bandwidth-trivial).
    {
        int n = ROWS * DIM;
        cvt_f32_to_f16<<<(n + 255) / 256, 256, 0, stream>>>(x, xh, n);
        n = QKVDIM * DIM;
        cvt_f32_to_f16<<<(n + 255) / 256, 256, 0, stream>>>(w_qkv, wqkvh, n);
        n = DIM * DIM;
        cvt_f32_to_f16<<<(n + 255) / 256, 256, 0, stream>>>(w_proj, wprjh, n);
    }

    // 2) QKV projection GEMM (8192 x 768 x 2304).
    qkv_gemm_kernel<<<dim3(ROWS / 32, QKVDIM / 256), 128, 0, stream>>>(
        xh, wqkvh, qbuf, kbuf, vbuf);

    // 3) Attention: one wave per (b, h, 16-row block); 64 KB dynamic LDS.
    attention_kernel<<<BATCH * NHEAD * (SEQ / 16), 32,
                       (size_t)16 * SEQ * sizeof(float), stream>>>(
        qbuf, kbuf, vbuf, ctx);

    // 4) Output projection GEMM (8192 x 768 x 768) + bias, f32 out.
    proj_gemm_kernel<<<dim3(ROWS / 32, DIM / 256), 128, 0, stream>>>(
        ctx, wprjh, b_proj, out);
}
